// QuantumNeuralNetwork_85074712199265
// MI455X (gfx1250) — matrix operand order repaired and checked by reading, not compiled, not run on hardware
//
#include <hip/hip_runtime.h>
#include <hip/hip_bf16.h>

// ---------------------------------------------------------------------------
// CDNA5 / gfx1250 fused QNN:
//   x(65536x256) @ W_in^T -> 4 angles -> 4-qubit circuit -> 4 evs
//   -> relu(@W1^T) 512 -> relu(@W2^T) 256 (bf16 WMMA) -> @W_out^T 10 (WMMA)
// One 256-thread block (8 wave32) handles 128 rows; h1/h2 live in LDS.
// W2/W_out pre-swizzled to bf16 WMMA B-operand layout in workspace.
// W2 staging: Tensor Data Mover (tensor_load_to_lds) double-buffered.
// ---------------------------------------------------------------------------

typedef __attribute__((ext_vector_type(16))) __bf16   v16bf;
typedef __attribute__((ext_vector_type(8)))  float    v8f;
typedef __attribute__((ext_vector_type(4)))  unsigned v4u;
typedef __attribute__((ext_vector_type(8)))  int      v8i;
typedef __attribute__((ext_vector_type(4)))  int      v4i;

#if __has_builtin(__builtin_amdgcn_tensor_load_to_lds) && \
    __has_builtin(__builtin_amdgcn_s_wait_tensorcnt)
#define QNN_TDM 1
#else
#define QNN_TDM 0
#endif

__device__ __forceinline__ unsigned short f2bf(float f) {
    union { float f; unsigned u; } cv;
    cv.f = f;
    unsigned u = cv.u;
    u += 0x7fffu + ((u >> 16) & 1u);   // round-to-nearest-even
    return (unsigned short)(u >> 16);
}

// A-operand (16-bit, 16x32) K swizzle: lanes0-15 hold K{0..7,16..23}, lanes16-31 K{8..15,24..31}
__device__ __forceinline__ void a_swz(int kk, int& hi, int& j) {
    hi = (kk >> 3) & 1;
    j  = (kk & 7) | ((kk & 16) >> 1);
}

#if QNN_TDM
// TDM: DMA one contiguous 8 KB chunk (2048 dwords as a 2048x1 tile) global->LDS.
// D# built per CDNA5 ISA 8.3/8.4: group0 {count, lds_addr, global_addr, type=2},
// group1 {data_size=4B, tensor_dim0=tile_dim0=2048, tensor_dim1=tile_dim1=1,
// stride0=2048}.  Groups 2/3 (and aux group) zero: dims 2..4 unused.
// Tracked by TENSORcnt; this toolchain's builtin takes 6 args.
__device__ __forceinline__ void tdm_copy_8kb(const unsigned short* src,
                                             unsigned short* dstLds) {
    unsigned long long ga = (unsigned long long)(uintptr_t)src;
    unsigned ldsoff = (unsigned)(uintptr_t)dstLds;   // flat LDS aperture: [31:0] = LDS offset
    v4u g0 = { 1u,                                   // count=1, no gather/iterate
               ldsoff,
               (unsigned)ga,
               (unsigned)((ga >> 32) & 0x1FFFFFFu) | (2u << 30) };  // type=2 "image"
    v8i g1 = { (int)(2u << 16),      // data_size code 2 = 4 bytes
               (int)(2048u << 16),   // tensor_dim0[15:0]=2048 at bits 63:48
               (int)(1u << 16),      // tensor_dim0[31:16]=0 | tensor_dim1[15:0]=1
               (int)(2048u << 16),   // tensor_dim1[31:16]=0 | tile_dim0=2048
               (int)1,               // tile_dim1=1, tile_dim2=0
               (int)2048,            // tensor_dim0_stride[31:0]=2048
               (int)(2048u << 16),   // stride0[47:32]=0 | stride1[15:0]=2048
               0 };                  // stride1[47:16]=0
    v4i gz4 = { 0, 0, 0, 0 };
    v8i gz8 = { 0, 0, 0, 0, 0, 0, 0, 0 };
    __builtin_amdgcn_tensor_load_to_lds(g0, g1, gz4, gz4, gz8, 0);
}
#endif

// ---------------------------------------------------------------------------
// Prep: swizzle W2 (256x512) and W_out (10x256, padded to 16) to bf16 B layout.
// B layout (32x16, 16-bit): element j of lane l is K = 32*s + (l>>4)*8 + (j&7) + (j>>3)*16,
// N = 16*ntile + (l&15).  ws index = ((s*16+nt)*32 + lane)*16 + j.
// ---------------------------------------------------------------------------
__global__ void __launch_bounds__(256)
qnn_prep(const float* __restrict__ W2, const float* __restrict__ W_out,
         unsigned short* __restrict__ wsW2, unsigned short* __restrict__ wsWout)
{
    int e = blockIdx.x * 256 + threadIdx.x;
    if (e < 131072) {                       // 16 s * 16 nt * 32 lanes * 16 j
        int j    = e & 15;
        int lane = (e >> 4) & 31;
        int nt   = (e >> 9) & 15;
        int s    = e >> 13;
        int n    = nt * 16 + (lane & 15);
        int k    = s * 32 + (lane >> 4) * 8 + (j & 7) + ((j >> 3) << 4);
        wsW2[e] = f2bf(W2[n * 512 + k]);
    } else if (e < 131072 + 4096) {         // 8 s * 32 lanes * 16 j
        int e2   = e - 131072;
        int j    = e2 & 15;
        int lane = (e2 >> 4) & 31;
        int s2   = e2 >> 9;
        int n    = lane & 15;
        int k    = s2 * 32 + (lane >> 4) * 8 + (j & 7) + ((j >> 3) << 4);
        wsWout[e2] = (n < 10) ? f2bf(W_out[n * 256 + k]) : (unsigned short)0;
    }
}

// LDS layout (bytes):
//   [0      , 131072) h1  bf16 swizzled  (8 mt x 16 s x 32 lane x 16 half)
//   [131072 , 196608) h2  bf16 swizzled  (8 mt x  8 s x 32 lane x 16 half)
//   [196608 , 212992) B stage, 2 x 4096 halves (8 nt x 32 lane x 16 half)
//   [212992 , 217088) q / partial dots, 256 x 4 floats
#define LDS_BYTES 217088

__global__ void __launch_bounds__(256)
qnn_main(const float* __restrict__ x,    const float* __restrict__ W_in,
         const float* __restrict__ b_in, const float* __restrict__ qw,
         const float* __restrict__ W1,   const float* __restrict__ b1,
         const float* __restrict__ b2,   const float* __restrict__ b_out,
         const unsigned short* __restrict__ wsW2,
         const unsigned short* __restrict__ wsWout,
         float* __restrict__ out)
{
    extern __shared__ __align__(128) char smem[];
    unsigned short* ldsH1  = (unsigned short*)smem;
    unsigned short* ldsH2  = (unsigned short*)(smem + 131072);
    unsigned short* ldsBst = (unsigned short*)(smem + 196608);
    float*          ldsQ   = (float*)(smem + 212992);

    const int t    = threadIdx.x;
    const int lane = t & 31;
    const int w    = t >> 5;          // wave id 0..7  == M-tile
    const int row0 = blockIdx.x * 128;

    // ---------------- Phase A1: input GEMM, split-K fp32 ----------------
    {
        const int r  = t & 127;
        const int kh = t >> 7;                           // K half 0/1
        const float4* x4  = (const float4*)(x + (size_t)(row0 + r) * 256 + kh * 128);
        const float4* w0p = (const float4*)(W_in +   0 + kh * 128);
        const float4* w1p = (const float4*)(W_in + 256 + kh * 128);
        const float4* w2p = (const float4*)(W_in + 512 + kh * 128);
        const float4* w3p = (const float4*)(W_in + 768 + kh * 128);
        float a0 = 0.f, a1 = 0.f, a2 = 0.f, a3 = 0.f;
        #pragma unroll 8
        for (int k4 = 0; k4 < 32; ++k4) {
            float4 xv = x4[k4];
            float4 wa = w0p[k4], wb = w1p[k4], wc = w2p[k4], wd = w3p[k4];
            a0 += xv.x*wa.x + xv.y*wa.y + xv.z*wa.z + xv.w*wa.w;
            a1 += xv.x*wb.x + xv.y*wb.y + xv.z*wb.z + xv.w*wb.w;
            a2 += xv.x*wc.x + xv.y*wc.y + xv.z*wc.z + xv.w*wc.w;
            a3 += xv.x*wd.x + xv.y*wd.y + xv.z*wd.z + xv.w*wd.w;
        }
        ldsQ[t*4+0] = a0; ldsQ[t*4+1] = a1; ldsQ[t*4+2] = a2; ldsQ[t*4+3] = a3;
    }
    __syncthreads();

    // ---------------- Phase A2: 4-qubit circuit (threads 0..127) --------
    if (t < 128) {
        float cc[4], ss[4];
        #pragma unroll
        for (int j = 0; j < 4; ++j) {
            float ang = 0.5f * (ldsQ[t*4+j] + ldsQ[(t+128)*4+j] + b_in[j]);
            sincosf(ang, &ss[j], &cc[j]);
        }
        // |psi> = tensor-product of (cos, sin); qubit i at bit (3-i)
        float ar[16], ai[16];
        #pragma unroll
        for (int i = 0; i < 16; ++i) {
            float p = ((i & 8) ? ss[0] : cc[0]) * ((i & 4) ? ss[1] : cc[1])
                    * ((i & 2) ? ss[2] : cc[2]) * ((i & 1) ? ss[3] : cc[3]);
            ar[i] = p; ai[i] = 0.f;
        }
        // U_q = Rz(w2)*Ry(w1)*Rx(w0), applied per qubit
        #pragma unroll
        for (int qb = 0; qb < 4; ++qb) {
            float sx, cx, sy, cy, sz, cz;
            sincosf(0.5f * qw[3*qb+0], &sx, &cx);
            sincosf(0.5f * qw[3*qb+1], &sy, &cy);
            sincosf(0.5f * qw[3*qb+2], &sz, &cz);
            float m00r =  cy*cx, m00i =  sy*sx;
            float m01r = -sy*cx, m01i = -cy*sx;
            float m10r =  sy*cx, m10i = -cy*sx;
            float m11r =  cy*cx, m11i = -sy*sx;
            // row0 *= (cz - i sz); row1 *= (cz + i sz)
            float u00r = cz*m00r + sz*m00i, u00i = cz*m00i - sz*m00r;
            float u01r = cz*m01r + sz*m01i, u01i = cz*m01i - sz*m01r;
            float u10r = cz*m10r - sz*m10i, u10i = cz*m10i + sz*m10r;
            float u11r = cz*m11r - sz*m11i, u11i = cz*m11i + sz*m11r;
            const int st = 1 << (3 - qb);
            #pragma unroll
            for (int b0 = 0; b0 < 16; ++b0) {
                if (b0 & st) continue;
                const int b1i = b0 | st;
                float xr = ar[b0], xi = ai[b0], yr = ar[b1i], yi = ai[b1i];
                ar[b0]  = u00r*xr - u00i*xi + u01r*yr - u01i*yi;
                ai[b0]  = u00r*xi + u00i*xr + u01r*yi + u01i*yr;
                ar[b1i] = u10r*xr - u10i*xi + u11r*yr - u11i*yi;
                ai[b1i] = u10r*xi + u10i*xr + u11r*yi + u11i*yr;
            }
        }
        // CNOT chain (0,1),(1,2),(2,3): control bit set -> swap target pair
        #pragma unroll
        for (int cq = 0; cq < 3; ++cq) {
            const int mc  = 1 << (3 - cq);
            const int mtb = 1 << (2 - cq);
            #pragma unroll
            for (int i = 0; i < 16; ++i) {
                if ((i & mc) && !(i & mtb)) {
                    const int j2 = i | mtb;
                    float tr = ar[i]; ar[i] = ar[j2]; ar[j2] = tr;
                    float ti = ai[i]; ai[i] = ai[j2]; ai[j2] = ti;
                }
            }
        }
        float pv[16];
        #pragma unroll
        for (int i = 0; i < 16; ++i) pv[i] = ar[i]*ar[i] + ai[i]*ai[i];
        #pragma unroll
        for (int qb = 0; qb < 4; ++qb) {
            const int m = 1 << (3 - qb);
            float ev = 0.f;
            #pragma unroll
            for (int i = 0; i < 16; ++i) ev += (i & m) ? -pv[i] : pv[i];
            ldsQ[t*4 + qb] = ev;
        }
    }
    __syncthreads();

    // ---------------- Phase B: q -> h1 (512, relu) -> LDS A-swizzle -----
    {
        const int n0 = 2 * t;                          // each thread: cols n0, n0+1
        float4 w1a = *(const float4*)(W1 + n0 * 4);
        float4 w1b = *(const float4*)(W1 + (n0 + 1) * 4);
        float bb0 = b1[n0], bb1 = b1[n0 + 1];
        int hi, j;  a_swz(n0 & 31, hi, j);             // j even; j,j+1 contiguous
        const int s = n0 >> 5;
        for (int it = 0; it < 128; ++it) {             // row = it for all threads
            float4 q4 = *(const float4*)(ldsQ + it * 4);
            float v0 = fmaxf(q4.x*w1a.x + q4.y*w1a.y + q4.z*w1a.z + q4.w*w1a.w + bb0, 0.f);
            float v1 = fmaxf(q4.x*w1b.x + q4.y*w1b.y + q4.z*w1b.z + q4.w*w1b.w + bb1, 0.f);
            unsigned pack = (unsigned)f2bf(v0) | ((unsigned)f2bf(v1) << 16);
            const int lane2 = (hi << 4) | (it & 15);
            const int off   = (((it >> 4) * 16 + s) * 32 + lane2) * 16 + j;  // halves
            *(unsigned*)(ldsH1 + off) = pack;
        }
    }
    __syncthreads();

    // ---------------- Phase C: h1(128x512) @ W2^T -> h2(128x256), WMMA --
    for (int p = 0; p < 2; ++p) {                       // N halves: cols p*128..+127
        v8f acc[8];
        #pragma unroll
        for (int nt = 0; nt < 8; ++nt)
            #pragma unroll
            for (int r = 0; r < 8; ++r) acc[nt][r] = 0.f;

#if QNN_TDM
        // ---- TDM double-buffered staging: wave 0 drives the DMA ----
        if (w == 0) {
            tdm_copy_8kb(wsW2 + (0 * 16 + p * 8) * 512, ldsBst);
            __builtin_amdgcn_s_wait_tensorcnt(0);
        }
        __syncthreads();
        for (int s = 0; s < 16; ++s) {
            if (w == 0 && s < 15)                       // DMA s+1 into other buffer
                tdm_copy_8kb(wsW2 + ((s + 1) * 16 + p * 8) * 512,
                             ldsBst + ((s + 1) & 1) * 4096);
            v16bf A = *(const v16bf*)(ldsH1 + ((w * 16 + s) * 32 + lane) * 16);
            const unsigned short* bbase = ldsBst + (s & 1) * 4096;
            #pragma unroll
            for (int nt = 0; nt < 8; ++nt) {
                v16bf B = *(const v16bf*)(bbase + (nt * 32 + lane) * 16);
                acc[nt] = __builtin_amdgcn_wmma_f32_16x16x32_bf16(
                              false, A, false, B, (short)0, acc[nt], false, false);
            }
            if (w == 0) __builtin_amdgcn_s_wait_tensorcnt(0);
            __syncthreads();
        }
#else
        // ---- Fallback: cooperative VGPR copy staging ----
        {
            const unsigned short* src = wsW2 + (0 * 16 + p * 8) * 512;
            *(uint4*)(ldsBst + t * 16)     = *(const uint4*)(src + t * 16);
            *(uint4*)(ldsBst + t * 16 + 8) = *(const uint4*)(src + t * 16 + 8);
        }
        __syncthreads();
        for (int s = 0; s < 16; ++s) {
            if (s < 14)
                __builtin_prefetch(wsW2 + ((s + 2) * 16 + p * 8) * 512 + t * 16, 0, 3);
            if (s < 15) {
                const unsigned short* src = wsW2 + ((s + 1) * 16 + p * 8) * 512;
                unsigned short* dst = ldsBst + ((s + 1) & 1) * 4096;
                *(uint4*)(dst + t * 16)     = *(const uint4*)(src + t * 16);
                *(uint4*)(dst + t * 16 + 8) = *(const uint4*)(src + t * 16 + 8);
            }
            v16bf A = *(const v16bf*)(ldsH1 + ((w * 16 + s) * 32 + lane) * 16);
            const unsigned short* bbase = ldsBst + (s & 1) * 4096;
            #pragma unroll
            for (int nt = 0; nt < 8; ++nt) {
                v16bf B = *(const v16bf*)(bbase + (nt * 32 + lane) * 16);
                acc[nt] = __builtin_amdgcn_wmma_f32_16x16x32_bf16(
                              false, A, false, B, (short)0, acc[nt], false, false);
            }
            __syncthreads();
        }
#endif

        {   // epilogue: bias + relu, store h2 in A-swizzle (K = h2 column)
            const int hi2  = lane >> 4;                 // C: lanes>=16 -> M=r+8
            const int nloc = lane & 15;
            #pragma unroll
            for (int nt = 0; nt < 8; ++nt) {
                const int ncol = p * 128 + nt * 16 + nloc;
                const float bias = b2[ncol];
                int hj, j;  a_swz(ncol & 31, hj, j);
                const int s2 = ncol >> 5;
                #pragma unroll
                for (int r = 0; r < 8; ++r) {
                    const int m = w * 16 + r + hi2 * 8;
                    float v = fmaxf(acc[nt][r] + bias, 0.f);
                    const int lane2 = (hj << 4) | (m & 15);
                    ldsH2[((w * 8 + s2) * 32 + lane2) * 16 + j] = f2bf(v);
                }
            }
        }
        __syncthreads();
    }

    // ---------------- Phase D: h2 @ W_out^T (N pad 16) + b_out ----------
    {
        v8f acc;
        #pragma unroll
        for (int r = 0; r < 8; ++r) acc[r] = 0.f;
        #pragma unroll
        for (int s2 = 0; s2 < 8; ++s2) {
            v16bf A = *(const v16bf*)(ldsH2 + ((w * 8 + s2) * 32 + lane) * 16);
            v16bf B = *(const v16bf*)(wsWout + (s2 * 32 + lane) * 16);
            acc = __builtin_amdgcn_wmma_f32_16x16x32_bf16(
                      false, A, false, B, (short)0, acc, false, false);
        }
        const int n = lane & 15;
        if (n < 10) {
            const float bias = b_out[n];
            #pragma unroll
            for (int r = 0; r < 8; ++r) {
                const int m = row0 + w * 16 + r + (lane >> 4) * 8;
                out[(size_t)m * 10 + n] = acc[r] + bias;
            }
        }
    }
}

extern "C" void kernel_launch(void* const* d_in, const int* in_sizes, int n_in,
                              void* d_out, int out_size, void* d_ws, size_t ws_size,
                              hipStream_t stream) {
    (void)in_sizes; (void)n_in; (void)out_size; (void)ws_size;
    const float* x     = (const float*)d_in[0];
    const float* W_in  = (const float*)d_in[1];
    const float* b_in  = (const float*)d_in[2];
    const float* qw    = (const float*)d_in[3];
    const float* W1    = (const float*)d_in[4];
    const float* b1    = (const float*)d_in[5];
    const float* W2    = (const float*)d_in[6];
    const float* b2    = (const float*)d_in[7];
    const float* W_out = (const float*)d_in[8];
    const float* b_out = (const float*)d_in[9];
    float* out = (float*)d_out;

    unsigned short* wsW2   = (unsigned short*)d_ws;
    unsigned short* wsWout = wsW2 + 131072;

    qnn_prep<<<(131072 + 4096 + 255) / 256, 256, 0, stream>>>(W2, W_out, wsW2, wsWout);

    (void)hipFuncSetAttribute((const void*)qnn_main,
                              hipFuncAttributeMaxDynamicSharedMemorySize, LDS_BYTES);
    qnn_main<<<512, 256, LDS_BYTES, stream>>>(x, W_in, b_in, qw, W1, b1, b2, b_out,
                                              wsW2, wsWout, out);
}
